// RegionLayer_9861244911999
// MI455X (gfx1250) — matrix operand order repaired and checked by reading, not compiled, not run on hardware
//
#include <hip/hip_runtime.h>

typedef __attribute__((ext_vector_type(16))) __bf16 v16bf;
typedef __attribute__((ext_vector_type(8)))  __bf16 v8bf;
typedef __attribute__((ext_vector_type(8)))  float  v8f;

#define GRID_I 8
#define GRID_J 8
#define NT     64        // tiles
#define B_     32
#define C_     32
#define H_     160
#define W_     160
#define TH     20
#define TW     20
#define KTOT   (C_ * 9)  // 288, permuted order k' = (kh*3+kw)*32 + ci
#define EPS_   1e-5f

// ---------------------------------------------------------------------------
// Kernel 1: per-(tile, channel) BN statistics over (B, th, tw) = 12800 elems.
// Writes scale = gamma * rsqrt(var+eps), shift = beta - mean*scale into ws.
// ---------------------------------------------------------------------------
__launch_bounds__(256)
__global__ void bn_stats_kernel(const float* __restrict__ x,
                                const float* __restrict__ gamma,
                                const float* __restrict__ beta,
                                float* __restrict__ scale,
                                float* __restrict__ shift) {
    const int t  = blockIdx.x;          // tile 0..63
    const int c  = blockIdx.y;          // channel 0..31
    const int ti = t / GRID_J, tj = t % GRID_J;

    __shared__ float s_sum[256];
    __shared__ float s_sq[256];

    float sum = 0.f, sq = 0.f;
    for (int e = threadIdx.x; e < B_ * TH * TW; e += 256) {
        const int b  = e / (TH * TW);
        const int p  = e % (TH * TW);
        const int gy = ti * TH + p / TW;
        const int gx = tj * TW + p % TW;
        const float v = x[((b * C_ + c) * H_ + gy) * W_ + gx];
        sum += v;
        sq  += v * v;
    }
    s_sum[threadIdx.x] = sum;
    s_sq[threadIdx.x]  = sq;
    __syncthreads();
    for (int s = 128; s > 0; s >>= 1) {
        if ((int)threadIdx.x < s) {
            s_sum[threadIdx.x] += s_sum[threadIdx.x + s];
            s_sq[threadIdx.x]  += s_sq[threadIdx.x + s];
        }
        __syncthreads();
    }
    if (threadIdx.x == 0) {
        const float n    = (float)(B_ * TH * TW);
        const float mean = s_sum[0] / n;
        const float var  = s_sq[0] / n - mean * mean;   // biased, matches torch BN
        const float sc   = gamma[t * C_ + c] * rsqrtf(var + EPS_);
        scale[t * C_ + c] = sc;
        shift[t * C_ + c] = beta[t * C_ + c] - mean * sc;
    }
}

// ---------------------------------------------------------------------------
// Kernel 2: per-(tile, batch) BN-apply + ReLU + 3x3 conv (implicit GEMM via
// V_WMMA_F32_16X16X32_BF16) + bias + residual.
//   GEMM: D[co, p] = sum_k' W[co, k'] * im2col[k', p], M=32, N=400, K=288,
//   K permuted as k' = (kh*3+kw)*32 + ci: each K=32 WMMA step is one conv tap
//   over all 32 input channels -> contiguous LDS fragments (ds_load_b128).
// Activations live in LDS channels-last: h[y][x][ci].
// A fragments (weights) are preloaded to registers once; B fragments are
// double-buffered so tap s+1 loads overlap tap s WMMAs.
// ---------------------------------------------------------------------------
__launch_bounds__(256)
__global__ void region_conv_kernel(const float* __restrict__ x,
                                   const float* __restrict__ conv_w,
                                   const float* __restrict__ conv_b,
                                   const float* __restrict__ scale,
                                   const float* __restrict__ shift,
                                   float* __restrict__ out) {
    const int t  = blockIdx.x;   // tile
    const int b  = blockIdx.y;   // batch
    const int ti = t / GRID_J, tj = t % GRID_J;

    __shared__ alignas(16) __bf16 h_lds[TH + 2][TW + 2][C_]; // zero halo, ch-last
    __shared__ alignas(16) __bf16 w_lds[C_][KTOT];           // [co][k']
    __shared__ float sc_lds[C_], sh_lds[C_], bias_lds[C_];

    const int tid = threadIdx.x;

    // zero the whole padded activation buffer (halo = SAME padding)
    for (int e = tid; e < (TH + 2) * (TW + 2) * C_; e += 256)
        ((__bf16*)h_lds)[e] = (__bf16)0.f;
    if (tid < C_) {
        sc_lds[tid]   = scale[t * C_ + tid];
        sh_lds[tid]   = shift[t * C_ + tid];
        bias_lds[tid] = conv_b[t * C_ + tid];
    }
    __syncthreads();

    // BN-apply + ReLU + f32->bf16, interior fill (channels-last in LDS)
    for (int e = tid; e < C_ * TH * TW; e += 256) {
        const int c  = e / (TH * TW);
        const int p  = e % (TH * TW);
        const int oy = p / TW, ox = p % TW;
        float v = x[((b * C_ + c) * H_ + ti * TH + oy) * W_ + tj * TW + ox];
        v = v * sc_lds[c] + sh_lds[c];
        v = v > 0.f ? v : 0.f;
        h_lds[oy + 1][ox + 1][c] = (__bf16)v;
    }
    // weights f32->bf16 with K permutation: src k = ci*9 + kh*3 + kw,
    // dst k' = (kh*3+kw)*32 + ci
    const float* wt = conv_w + t * C_ * KTOT;
    for (int e = tid; e < C_ * KTOT; e += 256) {
        const int co  = e / KTOT;
        const int r   = e % KTOT;         // ci*9 + kh*3 + kw
        const int ci  = r / 9;
        const int tap = r % 9;            // kh*3 + kw
        w_lds[co][tap * C_ + ci] = (__bf16)wt[e];
    }
    __syncthreads();

    const int lane = tid & 31;
    const int wave = tid >> 5;
    const int row  = lane & 15;   // M (A) / N (B,D) index within fragment
    const int half = lane >> 4;   // lane half selects K sub-range per ISA layout

    // ---- Preload all A fragments (loop-invariant across p-tiles) ----
    // A element e=0..7 -> K = half*8+e ; e=8..15 -> K = 16+half*8+(e-8)
    const __bf16* a0base = &w_lds[row][half * 8];         // co tile 0
    const __bf16* a1base = &w_lds[16 + row][half * 8];    // co tile 1
    v16bf a0f[9], a1f[9];
    #pragma unroll
    for (int s = 0; s < 9; ++s) {
        v8bf l0 = *(const v8bf*)(a0base + s * 32);
        v8bf h0 = *(const v8bf*)(a0base + s * 32 + 16);
        a0f[s] = __builtin_shufflevector(l0, h0,
                   0, 1, 2, 3, 4, 5, 6, 7, 8, 9, 10, 11, 12, 13, 14, 15);
        v8bf l1 = *(const v8bf*)(a1base + s * 32);
        v8bf h1 = *(const v8bf*)(a1base + s * 32 + 16);
        a1f[s] = __builtin_shufflevector(l1, h1,
                   0, 1, 2, 3, 4, 5, 6, 7, 8, 9, 10, 11, 12, 13, 14, 15);
    }

    // 25 p-tiles round-robin over 8 waves; each wave does both co-tiles so the
    // B (im2col) fragment is loaded once per WMMA pair.
    for (int nT = wave; nT < 25; nT += 8) {
        const int pcol = nT * 16 + row;           // this lane's output pixel
        const int oy   = pcol / TW, ox = pcol % TW;

        // Per-lane LDS base; tap offsets are compile-time immediates.
        const __bf16* bbase = &h_lds[oy][ox][half * 8];   // B: K == ci

        // ---- software-pipelined B fragments (2-deep double buffer) ----
        v8bf blo[2], bhi[2];
        blo[0] = *(const v8bf*)(bbase);           // tap 0: (kh=0,kw=0)
        bhi[0] = *(const v8bf*)(bbase + 16);

        v8f acc0 = {};
        v8f acc1 = {};
        #pragma unroll
        for (int s = 0; s < 9; ++s) {             // one (kh,kw) tap per step
            if (s < 8) {                          // prefetch next tap
                const int kh = (s + 1) / 3, kw = (s + 1) % 3;
                const int boff = (kh * (TW + 2) + kw) * C_;
                blo[(s + 1) & 1] = *(const v8bf*)(bbase + boff);
                bhi[(s + 1) & 1] = *(const v8bf*)(bbase + boff + 16);
            }
            v16bf bb = __builtin_shufflevector(blo[s & 1], bhi[s & 1],
                         0, 1, 2, 3, 4, 5, 6, 7, 8, 9, 10, 11, 12, 13, 14, 15);
            acc0 = __builtin_amdgcn_wmma_f32_16x16x32_bf16(
                       false, a0f[s], false, bb, (short)0, acc0, false, false);
            acc1 = __builtin_amdgcn_wmma_f32_16x16x32_bf16(
                       false, a1f[s], false, bb, (short)0, acc1, false, false);
        }

        // D layout: lanes 0-15 -> M = r, lanes 16-31 -> M = 8 + r; N = lane&15
        const int gy = ti * TH + oy;
        const int gx = tj * TW + ox;
        #pragma unroll
        for (int r = 0; r < 8; ++r) {
            const int co0  = half * 8 + r;        // co tile 0
            const int co1  = 16 + co0;            // co tile 1
            const int idx0 = ((b * C_ + co0) * H_ + gy) * W_ + gx;
            const int idx1 = ((b * C_ + co1) * H_ + gy) * W_ + gx;
            out[idx0] = acc0[r] + bias_lds[co0] + x[idx0];
            out[idx1] = acc1[r] + bias_lds[co1] + x[idx1];
        }
    }
}

// ---------------------------------------------------------------------------
extern "C" void kernel_launch(void* const* d_in, const int* in_sizes, int n_in,
                              void* d_out, int out_size, void* d_ws, size_t ws_size,
                              hipStream_t stream) {
    const float* x      = (const float*)d_in[0];
    const float* gamma  = (const float*)d_in[1];
    const float* beta   = (const float*)d_in[2];
    const float* conv_w = (const float*)d_in[3];
    const float* conv_b = (const float*)d_in[4];
    float* out = (float*)d_out;

    float* scale = (float*)d_ws;           // NT*C_ floats
    float* shift = scale + NT * C_;        // NT*C_ floats

    dim3 g1(NT, C_);
    bn_stats_kernel<<<g1, 256, 0, stream>>>(x, gamma, beta, scale, shift);

    dim3 g2(NT, B_);
    region_conv_kernel<<<g2, 256, 0, stream>>>(x, conv_w, conv_b, scale, shift, out);
}
